// EncoderSimilarity_2027224563759
// MI455X (gfx1250) — matrix operand order repaired and checked by reading, not compiled, not run on hardware
//
#include <hip/hip_runtime.h>
#include <hip/hip_bf16.h>

// ---------------------------------------------------------------------------
// SGRAF-style EncoderSimilarity forward for MI455X (gfx1250, wave32, WMMA).
// All GEMMs: v_wmma_f32_16x16x32_bf16 with a 32x32 macro-tile per wave
// (2x2 fragment reuse -> 2 b128 loads per WMMA), guard-free vectorized
// fragment loads (weights pre-transposed so B loads are contiguous per lane);
// f32 VALU for softmax / l2norm / activations / reductions.
// ---------------------------------------------------------------------------

typedef unsigned short u16t;
typedef __attribute__((ext_vector_type(16))) __bf16 v16bf;
typedef __attribute__((ext_vector_type(8)))  __bf16 v8bf;
typedef __attribute__((ext_vector_type(8)))  float  v8f;

constexpr int Bn = 48;    // images
constexpr int Rr = 36;    // regions per image
constexpr int Ll = 40;    // words per caption
constexpr int Ed = 1024;  // embedding dim
constexpr int Sd = 256;   // similarity dim
constexpr int Td = 3;     // SGR steps
constexpr int Cn = 48;    // captions
constexpr int Nn = Ll + 1;        // 41 graph nodes (glo + words)
constexpr int NnP = 64;           // K-padded node count for edge@se GEMM
constexpr int CC = 8;             // captions per chunk
constexpr int NCH = Cn / CC;      // 6 chunks
constexpr float BN_S = 0.9999950000374997f; // 1/sqrt(1+1e-5), eval BatchNorm

// ------------------------------- helpers -----------------------------------
__device__ __forceinline__ u16t f2bf(float f) {
  unsigned u = __builtin_bit_cast(unsigned, f);
  u += 0x7FFFu + ((u >> 16) & 1u);           // round-to-nearest-even
  return (u16t)(u >> 16);
}
__device__ __forceinline__ float bf2f(u16t u) {
  return __builtin_bit_cast(float, (unsigned)u << 16);
}
__device__ __forceinline__ __bf16 u2bf(u16t u) {
  return __builtin_bit_cast(__bf16, u);
}

// A-fragment: two contiguous 8-elt bf16 runs:
//   lane half h: element j -> K=h*16+j
__device__ __forceinline__ v16bf load_a_frag(const u16t* Arow, int kk) {
  const v8bf a0 = *(const v8bf*)(Arow + kk);
  const v8bf a1 = *(const v8bf*)(Arow + kk + 8);
  return __builtin_shufflevector(a0, a1,
      0, 1, 2, 3, 4, 5, 6, 7, 8, 9, 10, 11, 12, 13, 14, 15);
}
// B-fragment, BT (NxK row-major): lane group h holds K=kk+16h..kk+16h+15
__device__ __forceinline__ v16bf load_bt_frag(const u16t* Bp, int kk) {
  const v8bf b0 = *(const v8bf*)(Bp + kk);
  const v8bf b1 = *(const v8bf*)(Bp + kk + 8);
  return __builtin_shufflevector(b0, b1,
      0, 1, 2, 3, 4, 5, 6, 7, 8, 9, 10, 11, 12, 13, 14, 15);
}

// ------------------------------ WMMA GEMM ----------------------------------
// C[M,N] = act( A[M,K](bf16,row-major) x B + bias[n] ), optional batch strides.
// One wave computes a 32x32 C macro-tile (2x2 WMMA tiles, fragment reuse).
// BT=true : B is NxK row-major (contiguous per-lane loads).
// BT=false: B is KxN row-major (strided per-lane loads; only edge@se GEMM).
// K must be a multiple of 32 (operands zero-padded in K where needed).
// No bounds checks on loads: operands live in workspace with tail slack, and
// out-of-range tile rows/cols are never stored. EXEC stays full in the K loop.
// ACT: 0 none, 1 tanh(scale*x), 2 leaky_relu(0.1), 3 relu.
template<int ACT, bool BT, bool OUTBF>
__global__ __launch_bounds__(256)
void wmma_gemm(const u16t* __restrict__ A, long sA, int lda,
               const u16t* __restrict__ Bm, long sB, int ldb,
               void* __restrict__ Cm, long sC, int ldc,
               const float* __restrict__ bias,
               int M, int N, int K, int tilesN, float scale)
{
  const int wave = threadIdx.x >> 5;
  const int lane = threadIdx.x & 31;
  const int tile = blockIdx.x * 8 + wave;    // 32x32 macro-tile id
  const int tm = tile / tilesN;
  const int tn = tile - tm * tilesN;
  const int m0 = tm * 32, n0 = tn * 32;
  if (m0 >= M) return;                 // uniform per wave -> EXEC full for WMMA
  const int batch = blockIdx.y;
  const int half = lane >> 4;          // lane group 0/1
  const int l15 = lane & 15;

  const u16t* Ab = A + (long)batch * sA;
  const u16t* Bb = Bm + (long)batch * sB;
  const u16t* Arow0 = Ab + (long)(m0 + l15) * lda + half * 16;
  const u16t* Arow1 = Arow0 + (long)16 * lda;
  const u16t* Bp0 = BT ? (Bb + (long)(n0 + l15) * ldb + half * 16)
                       : (Bb + (n0 + l15));
  const u16t* Bp1 = BT ? (Bp0 + (long)16 * ldb) : (Bp0 + 16);

  v8f acc00 = {}, acc01 = {}, acc10 = {}, acc11 = {};
#pragma unroll 2
  for (int kk = 0; kk < K; kk += 32) {
    const v16bf a0 = load_a_frag(Arow0, kk);
    const v16bf a1 = load_a_frag(Arow1, kk);
    v16bf b0, b1;
    if (BT) {
      b0 = load_bt_frag(Bp0, kk);
      b1 = load_bt_frag(Bp1, kk);
    } else {
#pragma unroll
      for (int j = 0; j < 16; ++j) {
        const long ko = (long)(kk + half * 16 + j) * ldb;
        b0[j] = u2bf(Bp0[ko]);
        b1[j] = u2bf(Bp1[ko]);
      }
    }
    acc00 = __builtin_amdgcn_wmma_f32_16x16x32_bf16(false, a0, false, b0,
                                                    (short)0, acc00, false, false);
    acc01 = __builtin_amdgcn_wmma_f32_16x16x32_bf16(false, a0, false, b1,
                                                    (short)0, acc01, false, false);
    acc10 = __builtin_amdgcn_wmma_f32_16x16x32_bf16(false, a1, false, b0,
                                                    (short)0, acc10, false, false);
    acc11 = __builtin_amdgcn_wmma_f32_16x16x32_bf16(false, a1, false, b1,
                                                    (short)0, acc11, false, false);
  }
  // C/D layout: lane group selects M half; vgpr v -> M = half*8 + v, N = lane&15
  const v8f* accs[4] = { &acc00, &acc01, &acc10, &acc11 };
#pragma unroll
  for (int mt = 0; mt < 2; ++mt) {
#pragma unroll
    for (int nt = 0; nt < 2; ++nt) {
      const int ncol = n0 + nt * 16 + l15;
      if (ncol >= N) continue;
      const float bv = bias ? bias[ncol] : 0.0f;
      const v8f& acc = *accs[mt * 2 + nt];
#pragma unroll
      for (int v = 0; v < 8; ++v) {
        const int m = m0 + mt * 16 + half * 8 + v;
        if (m < M) {
          float x = acc[v] + bv;
          if (ACT == 1)      x = tanhf(scale * x);
          else if (ACT == 2) x = x > 0.0f ? x : 0.1f * x;
          else if (ACT == 3) x = fmaxf(x, 0.0f);
          const long idx = (long)batch * sC + (long)m * ldc + ncol;
          if (OUTBF) ((u16t*)Cm)[idx] = f2bf(x);
          else       ((float*)Cm)[idx] = x;
        }
      }
    }
  }
}

template<int ACT, bool BT, bool OUTBF>
static void gemm(hipStream_t st, const u16t* A, long sA, int lda,
                 const u16t* B, long sB, int ldb,
                 void* C, long sC, int ldc, const float* bias,
                 int M, int N, int K, int batch, float scale = 1.0f)
{
  const int tilesM = (M + 31) / 32, tilesN = (N + 31) / 32;
  const int tiles = tilesM * tilesN;
  dim3 g((unsigned)((tiles + 7) / 8), (unsigned)batch, 1);
  wmma_gemm<ACT, BT, OUTBF><<<g, 256, 0, st>>>(A, sA, lda, B, sB, ldb,
                                               C, sC, ldc, bias, M, N, K,
                                               tilesN, scale);
}

// --------------------------- elementwise kernels ---------------------------
__global__ void to_bf16(const float* __restrict__ in, u16t* __restrict__ out, long n) {
  long i = blockIdx.x * 256L + threadIdx.x;
  if (i < n) out[i] = f2bf(in[i]);
}

// in: KxN row-major f32 -> out: NxK row-major bf16 (weight pre-transpose)
__global__ void to_bf16_t(const float* __restrict__ in, u16t* __restrict__ out,
                          int K, int N) {
  long i = blockIdx.x * 256L + threadIdx.x;
  if (i >= (long)K * N) return;
  int n = (int)(i % N); long k = i / N;
  out[(long)n * K + k] = f2bf(in[i]);
}

__global__ void mean_mid(const float* __restrict__ in, float* __restrict__ out,
                         int X, int Rm) {
  int i = blockIdx.x * 256 + threadIdx.x;
  if (i >= X * Ed) return;
  int x = i / Ed, e = i - x * Ed;
  float s = 0.0f;
  for (int r = 0; r < Rm; ++r) s += in[((long)x * Rm + r) * Ed + e];
  out[i] = s * (1.0f / (float)Rm);
}

// logits[row] = sum_e l[row,e]*g[row/grpsz,e]*wc[e] + bc
__global__ __launch_bounds__(256)
void sa_logits(const float* __restrict__ lmat, const float* __restrict__ gmat,
               const float* __restrict__ wc, const float* __restrict__ bc,
               float* __restrict__ out, int rowsPerGrp)
{
  const int row = blockIdx.x;
  const int grp = row / rowsPerGrp;
  __shared__ float red[256];
  const float* lp = lmat + (long)row * Ed;
  const float* gp = gmat + (long)grp * Ed;
  float s = 0.0f;
  for (int e = threadIdx.x; e < Ed; e += 256) s += lp[e] * gp[e] * wc[e];
  red[threadIdx.x] = s; __syncthreads();
  for (int o = 128; o > 0; o >>= 1) {
    if (threadIdx.x < o) red[threadIdx.x] += red[threadIdx.x + o];
    __syncthreads();
  }
  if (threadIdx.x == 0) out[row] = red[0] + bc[0];
}

// softmax over Rm logits, weighted sum of emb rows, l2norm -> out[g, Ed]
__global__ __launch_bounds__(256)
void softmax_pool_l2(const float* __restrict__ logits, const float* __restrict__ emb,
                     float* __restrict__ out, int Rm)
{
  const int g = blockIdx.x, t = threadIdx.x;
  __shared__ float w[64];
  __shared__ float red[256];
  if (t == 0) {
    float mx = -1e30f;
    for (int r = 0; r < Rm; ++r) mx = fmaxf(mx, logits[g * Rm + r]);
    float s = 0.0f;
    for (int r = 0; r < Rm; ++r) { float ex = __expf(logits[g * Rm + r] - mx); w[r] = ex; s += ex; }
    const float inv = 1.0f / s;
    for (int r = 0; r < Rm; ++r) w[r] *= inv;
  }
  __syncthreads();
  float sq = 0.0f;
  for (int e = t; e < Ed; e += 256) {
    float a = 0.0f;
    for (int r = 0; r < Rm; ++r) a += w[r] * emb[((long)g * Rm + r) * Ed + e];
    out[(long)g * Ed + e] = a; sq += a * a;
  }
  red[t] = sq; __syncthreads();
  for (int o = 128; o > 0; o >>= 1) { if (t < o) red[t] += red[t + o]; __syncthreads(); }
  const float inv = 1.0f / (sqrtf(red[0]) + 1e-8f);
  for (int e = t; e < Ed; e += 256) out[(long)g * Ed + e] *= inv;
}

__global__ void dglo_k(const float* __restrict__ ig, const float* __restrict__ cg,
                       u16t* __restrict__ out) {
  long i = blockIdx.x * 256L + threadIdx.x;
  if (i >= (long)Cn * Bn * Ed) return;
  int e = (int)(i % Ed); long t2 = i / Ed;
  int b = (int)(t2 % Bn); int c = (int)(t2 / Bn);
  float d = ig[(long)b * Ed + e] - cg[(long)c * Ed + e];
  out[i] = f2bf(d * d);
}

__global__ void row_l2norm(float* __restrict__ x, long rows, int w) {
  long r = blockIdx.x * 256L + threadIdx.x;
  if (r >= rows) return;
  float* p = x + r * (long)w;
  float s = 0.0f;
  for (int i = 0; i < w; ++i) s += p[i] * p[i];
  const float inv = 1.0f / (sqrtf(s) + 1e-8f);
  for (int i = 0; i < w; ++i) p[i] *= inv;
}

// a: (C*B, R, L) l2-normed over L; write aw[c,b,l,r] = softmax_r(9*a[c,b,r,l])
__global__ void attn_softmax_t(const float* __restrict__ a, float* __restrict__ aw) {
  long i = blockIdx.x * 256L + threadIdx.x;
  if (i >= (long)Cn * Bn * Ll) return;
  int l = (int)(i % Ll); long g = i / Ll;
  const float* ap = a + (g * Rr) * (long)Ll + l;
  float mx = -1e30f;
  for (int r = 0; r < Rr; ++r) mx = fmaxf(mx, ap[(long)r * Ll]);
  float s = 0.0f;
  for (int r = 0; r < Rr; ++r) s += __expf(9.0f * (ap[(long)r * Ll] - mx));
  const float inv = 1.0f / s;
  float* op = aw + (g * Ll + l) * (long)Rr;
  for (int r = 0; r < Rr; ++r) op[r] = __expf(9.0f * (ap[(long)r * Ll] - mx)) * inv;
}

// per (cc,b,l): wctx row = attn_w . img regions; l2norm; (wctx-cap)^2 -> bf16
__global__ __launch_bounds__(256)
void wctx_dloc(const float* __restrict__ aw, const float* __restrict__ img,
               const float* __restrict__ cap, u16t* __restrict__ dloc, int c0)
{
  const int row = blockIdx.x;                // over CC*Bn*Ll
  const int l = row % Ll; int t2 = row / Ll;
  const int b = t2 % Bn; const int cc = t2 / Bn;
  const int c = c0 + cc;
  __shared__ float w[Rr];
  __shared__ float red[256];
  const int t = threadIdx.x;
  if (t < Rr) w[t] = aw[(((long)c * Bn + b) * Ll + l) * Rr + t];
  __syncthreads();
  float u[4]; float sq = 0.0f;
#pragma unroll
  for (int i = 0; i < 4; ++i) {
    const int e = t + i * 256;
    float a = 0.0f;
    for (int r = 0; r < Rr; ++r) a += w[r] * img[((long)b * Rr + r) * Ed + e];
    u[i] = a; sq += a * a;
  }
  red[t] = sq; __syncthreads();
  for (int o = 128; o > 0; o >>= 1) { if (t < o) red[t] += red[t + o]; __syncthreads(); }
  const float inv = 1.0f / (sqrtf(red[0]) + 1e-8f);
  const float* cp = cap + ((long)c * Ll + l) * Ed;
  u16t* dp = dloc + (long)row * Ed;
#pragma unroll
  for (int i = 0; i < 4; ++i) {
    const int e = t + i * 256;
    const float d = u[i] * inv - cp[e];
    dp[e] = f2bf(d * d);
  }
}

// se[(cc,b), n=0] = sim_glo row (pre-normed); n>=1: l2norm(sim_loc row) -> bf16
__global__ void build_se(const float* __restrict__ simglo, const float* __restrict__ simloc,
                         u16t* __restrict__ se, int c0)
{
  int row = blockIdx.x * 256 + threadIdx.x;
  if (row >= CC * Bn * Nn) return;
  int n = row % Nn; int t2 = row / Nn; int b = t2 % Bn; int cc = t2 / Bn;
  u16t* op = se + (long)row * Sd;
  if (n == 0) {
    const float* sp = simglo + ((long)(c0 + cc) * Bn + b) * Sd;
    for (int s = 0; s < Sd; ++s) op[s] = f2bf(sp[s]);
  } else {
    const float* sp = simloc + (((long)cc * Bn + b) * Ll + (n - 1)) * Sd;
    float ss = 0.0f;
    for (int s = 0; s < Sd; ++s) ss += sp[s] * sp[s];
    const float inv = 1.0f / (sqrtf(ss) + 1e-8f);
    for (int s = 0; s < Sd; ++s) op[s] = f2bf(sp[s] * inv);
  }
}

// softmax over Nn cols, write bf16 rows padded with zeros to NnP (K padding)
__global__ void edge_softmax(const float* __restrict__ e, u16t* __restrict__ o) {
  long r = blockIdx.x * 256L + threadIdx.x;
  if (r >= (long)CC * Bn * Nn) return;
  const float* p = e + r * Nn;
  u16t* q = o + r * NnP;
  float mx = -1e30f;
  for (int i = 0; i < Nn; ++i) mx = fmaxf(mx, p[i]);
  float s = 0.0f;
  for (int i = 0; i < Nn; ++i) s += __expf(p[i] - mx);
  const float inv = 1.0f / s;
  for (int i = 0; i < Nn; ++i) q[i] = f2bf(__expf(p[i] - mx) * inv);
  for (int i = Nn; i < NnP; ++i) q[i] = 0;
}

__global__ void eval_k(const u16t* __restrict__ se, const float* __restrict__ wev,
                       const float* __restrict__ bev, float* __restrict__ out, int c0)
{
  int i = blockIdx.x * 256 + threadIdx.x;   // over CC*Bn, i == cc*Bn + b
  if (i >= CC * Bn) return;
  int b = i % Bn, cc = i / Bn;
  const u16t* p = se + (long)i * Nn * Sd;   // node 0 row
  float s = 0.0f;
  for (int j = 0; j < Sd; ++j) s += bf2f(p[j]) * wev[j];
  s += bev[0];
  out[(long)b * Cn + (c0 + cc)] = 1.0f / (1.0f + __expf(-s));  // sim.T layout
}

// ------------------------------- driver ------------------------------------
extern "C" void kernel_launch(void* const* d_in, const int* in_sizes, int n_in,
                              void* d_out, int out_size, void* d_ws, size_t ws_size,
                              hipStream_t stream)
{
  (void)in_sizes; (void)n_in; (void)out_size; (void)ws_size;
  const float* img    = (const float*)d_in[0];
  const float* cap    = (const float*)d_in[1];
  // d_in[2] = cap_lens (all full length) — unused, like the reference
  const float* vsa_wl = (const float*)d_in[3];
  const float* vsa_bl = (const float*)d_in[4];
  const float* vsa_wg = (const float*)d_in[5];
  const float* vsa_bg = (const float*)d_in[6];
  const float* vsa_wc = (const float*)d_in[7];
  const float* vsa_bc = (const float*)d_in[8];
  const float* tsa_wl = (const float*)d_in[9];
  const float* tsa_bl = (const float*)d_in[10];
  const float* tsa_wg = (const float*)d_in[11];
  const float* tsa_bg = (const float*)d_in[12];
  const float* tsa_wc = (const float*)d_in[13];
  const float* tsa_bc = (const float*)d_in[14];
  const float* w_loc  = (const float*)d_in[15];
  const float* b_loc  = (const float*)d_in[16];
  const float* w_glo  = (const float*)d_in[17];
  const float* b_glo  = (const float*)d_in[18];
  const float* w_ev   = (const float*)d_in[19];
  const float* b_ev   = (const float*)d_in[20];
  const float* sgr_wq = (const float*)d_in[21];
  const float* sgr_bq = (const float*)d_in[22];
  const float* sgr_wk = (const float*)d_in[23];
  const float* sgr_bk = (const float*)d_in[24];
  const float* sgr_wg = (const float*)d_in[25];
  const float* sgr_bg = (const float*)d_in[26];
  float* out = (float*)d_out;

  char* base = (char*)d_ws; size_t off = 0;
  auto alloc = [&](size_t bytes) -> void* {
    off = (off + 255) & ~(size_t)255;
    void* p = base + off; off += bytes; return p;
  };

  // persistent (~66 MB) — weight copies are stored TRANSPOSED (NxK) as bf16
  u16t* img_bf   = (u16t*)alloc((size_t)Bn * Rr * Ed * 2);
  u16t* cap_bf   = (u16t*)alloc((size_t)Cn * Ll * Ed * 2);
  u16t* vwlT_bf  = (u16t*)alloc((size_t)Ed * Ed * 2);
  u16t* vwgT_bf  = (u16t*)alloc((size_t)Ed * Ed * 2);
  u16t* twlT_bf  = (u16t*)alloc((size_t)Ed * Ed * 2);
  u16t* twgT_bf  = (u16t*)alloc((size_t)Ed * Ed * 2);
  u16t* wlocT_bf = (u16t*)alloc((size_t)Sd * Ed * 2);
  u16t* wgloT_bf = (u16t*)alloc((size_t)Sd * Ed * 2);
  u16t* wqT_bf   = (u16t*)alloc((size_t)Td * Sd * Sd * 2);
  u16t* wkT_bf   = (u16t*)alloc((size_t)Td * Sd * Sd * 2);
  u16t* wggT_bf  = (u16t*)alloc((size_t)Td * Sd * Sd * 2);
  float* img_ave = (float*)alloc((size_t)Bn * Ed * 4);
  u16t*  imga_bf = (u16t*)alloc((size_t)Bn * Ed * 2);
  float* cap_ave = (float*)alloc((size_t)Cn * Ed * 4);
  u16t*  capa_bf = (u16t*)alloc((size_t)Cn * Ed * 2);
  float* l_v     = (float*)alloc((size_t)Bn * Rr * Ed * 4);
  float* g_v     = (float*)alloc((size_t)Bn * Ed * 4);
  float* l_t     = (float*)alloc((size_t)Cn * Ll * Ed * 4);
  float* g_t     = (float*)alloc((size_t)Cn * Ed * 4);
  float* wv_log  = (float*)alloc((size_t)Bn * Rr * 4);
  float* wt_log  = (float*)alloc((size_t)Cn * Ll * 4);
  float* img_glo = (float*)alloc((size_t)Bn * Ed * 4);
  float* cap_glo = (float*)alloc((size_t)Cn * Ed * 4);
  u16t*  dglo_bf = (u16t*)alloc((size_t)Cn * Bn * Ed * 2);
  float* sim_glo = (float*)alloc((size_t)Cn * Bn * Sd * 4);
  float* attn_r  = (float*)alloc((size_t)Cn * Bn * Rr * Ll * 4);
  float* attn_w  = (float*)alloc((size_t)Cn * Bn * Ll * Rr * 4);
  // chunk (~84 MB)
  u16t*  dloc_bf = (u16t*)alloc((size_t)CC * Bn * Ll * Ed * 2);
  float* simloc  = (float*)alloc((size_t)CC * Bn * Ll * Sd * 4);
  u16t*  se_bf   = (u16t*)alloc((size_t)CC * Bn * Nn * Sd * 2);
  u16t*  q_bf    = (u16t*)alloc((size_t)CC * Bn * Nn * Sd * 2);
  u16t*  k_bf    = (u16t*)alloc((size_t)CC * Bn * Nn * Sd * 2);
  float* edge    = (float*)alloc((size_t)CC * Bn * Nn * Nn * 4);
  u16t*  edge_bf = (u16t*)alloc((size_t)CC * Bn * Nn * NnP * 2);  // K-padded rows
  u16t*  agg_bf  = (u16t*)alloc((size_t)CC * Bn * Nn * Sd * 2);
  (void)alloc((size_t)1 << 20);  // tail slack: absorbs guard-free tile overreads

  auto cv = [&](const float* src, u16t* dst, long n) {
    to_bf16<<<dim3((unsigned)((n + 255) / 256)), 256, 0, stream>>>(src, dst, n);
  };
  auto cvt = [&](const float* src, u16t* dst, int K, int N) {  // transpose KxN -> NxK
    long n = (long)K * N;
    to_bf16_t<<<dim3((unsigned)((n + 255) / 256)), 256, 0, stream>>>(src, dst, K, N);
  };
  cv(img, img_bf, (long)Bn * Rr * Ed);
  cv(cap, cap_bf, (long)Cn * Ll * Ed);
  cvt(vsa_wl, vwlT_bf, Ed, Ed);
  cvt(vsa_wg, vwgT_bf, Ed, Ed);
  cvt(tsa_wl, twlT_bf, Ed, Ed);
  cvt(tsa_wg, twgT_bf, Ed, Ed);
  cvt(w_loc, wlocT_bf, Ed, Sd);
  cvt(w_glo, wgloT_bf, Ed, Sd);
  for (int t = 0; t < Td; ++t) {
    cvt(sgr_wq + (long)t * Sd * Sd, wqT_bf + (long)t * Sd * Sd, Sd, Sd);
    cvt(sgr_wk + (long)t * Sd * Sd, wkT_bf + (long)t * Sd * Sd, Sd, Sd);
    cvt(sgr_wg + (long)t * Sd * Sd, wggT_bf + (long)t * Sd * Sd, Sd, Sd);
  }

  mean_mid<<<dim3((Bn * Ed + 255) / 256), 256, 0, stream>>>(img, img_ave, Bn, Rr);
  cv(img_ave, imga_bf, (long)Bn * Ed);
  mean_mid<<<dim3((Cn * Ed + 255) / 256), 256, 0, stream>>>(cap, cap_ave, Cn, Ll);
  cv(cap_ave, capa_bf, (long)Cn * Ed);

  // ---- VisualSA -> img_glo ----
  gemm<1, true, false>(stream, img_bf, 0, Ed, vwlT_bf, 0, Ed, l_v, 0, Ed,
                       vsa_bl, Bn * Rr, Ed, Ed, 1, BN_S);
  gemm<1, true, false>(stream, imga_bf, 0, Ed, vwgT_bf, 0, Ed, g_v, 0, Ed,
                       vsa_bg, Bn, Ed, Ed, 1, BN_S);
  sa_logits<<<dim3(Bn * Rr), 256, 0, stream>>>(l_v, g_v, vsa_wc, vsa_bc, wv_log, Rr);
  softmax_pool_l2<<<dim3(Bn), 256, 0, stream>>>(wv_log, img, img_glo, Rr);

  // ---- TextSA -> cap_glo ----
  gemm<1, true, false>(stream, cap_bf, 0, Ed, twlT_bf, 0, Ed, l_t, 0, Ed,
                       tsa_bl, Cn * Ll, Ed, Ed, 1, 1.0f);
  gemm<1, true, false>(stream, capa_bf, 0, Ed, twgT_bf, 0, Ed, g_t, 0, Ed,
                       tsa_bg, Cn, Ed, Ed, 1, 1.0f);
  sa_logits<<<dim3(Cn * Ll), 256, 0, stream>>>(l_t, g_t, tsa_wc, tsa_bc, wt_log, Ll);
  softmax_pool_l2<<<dim3(Cn), 256, 0, stream>>>(wt_log, cap, cap_glo, Ll);

  // ---- global similarity embedding (all caption-image pairs) ----
  dglo_k<<<dim3((unsigned)(((long)Cn * Bn * Ed + 255) / 256)), 256, 0, stream>>>(
      img_glo, cap_glo, dglo_bf);
  gemm<0, true, false>(stream, dglo_bf, 0, Ed, wgloT_bf, 0, Ed, sim_glo, 0, Sd,
                       b_glo, Cn * Bn, Sd, Ed, 1);
  row_l2norm<<<dim3((Cn * Bn + 255) / 256), 256, 0, stream>>>(sim_glo, (long)Cn * Bn, Sd);

  // ---- SCAN cross attention (batched over captions): a = leaky(img . cap^T) ----
  gemm<2, true, false>(stream, img_bf, 0, Ed, cap_bf, (long)Ll * Ed, Ed,
                       attn_r, (long)Bn * Rr * Ll, Ll, nullptr, Bn * Rr, Ll, Ed, Cn);
  row_l2norm<<<dim3((unsigned)(((long)Cn * Bn * Rr + 255) / 256)), 256, 0, stream>>>(
      attn_r, (long)Cn * Bn * Rr, Ll);
  attn_softmax_t<<<dim3((unsigned)(((long)Cn * Bn * Ll + 255) / 256)), 256, 0, stream>>>(
      attn_r, attn_w);

  // ---- per-chunk: sim_loc, SGR steps, eval ----
  for (int ch = 0; ch < NCH; ++ch) {
    const int c0 = ch * CC;
    wctx_dloc<<<dim3(CC * Bn * Ll), 256, 0, stream>>>(attn_w, img, cap, dloc_bf, c0);
    gemm<0, true, false>(stream, dloc_bf, 0, Ed, wlocT_bf, 0, Ed, simloc, 0, Sd,
                         b_loc, CC * Bn * Ll, Sd, Ed, 1);
    build_se<<<dim3((CC * Bn * Nn + 255) / 256), 256, 0, stream>>>(sim_glo, simloc, se_bf, c0);
    for (int t = 0; t < Td; ++t) {
      gemm<0, true, true>(stream, se_bf, 0, Sd, wqT_bf + (long)t * Sd * Sd, 0, Sd,
                          q_bf, 0, Sd, sgr_bq + (long)t * Sd, CC * Bn * Nn, Sd, Sd, 1);
      gemm<0, true, true>(stream, se_bf, 0, Sd, wkT_bf + (long)t * Sd * Sd, 0, Sd,
                          k_bf, 0, Sd, sgr_bk + (long)t * Sd, CC * Bn * Nn, Sd, Sd, 1);
      // edge = q . k^T, batched per (cc,b)
      gemm<0, true, false>(stream, q_bf, (long)Nn * Sd, Sd, k_bf, (long)Nn * Sd, Sd,
                           edge, (long)Nn * Nn, Nn, nullptr, Nn, Nn, Sd, CC * Bn);
      edge_softmax<<<dim3((CC * Bn * Nn + 255) / 256), 256, 0, stream>>>(edge, edge_bf);
      // agg = softmax(edge) . se, batched; K padded 41 -> 64 with zeros in A
      gemm<0, false, true>(stream, edge_bf, (long)Nn * NnP, NnP, se_bf, (long)Nn * Sd, Sd,
                           agg_bf, (long)Nn * Sd, Sd, nullptr, Nn, Sd, NnP, CC * Bn);
      // se = relu(agg . wg + bg)
      gemm<3, true, true>(stream, agg_bf, 0, Sd, wggT_bf + (long)t * Sd * Sd, 0, Sd,
                          se_bf, 0, Sd, sgr_bg + (long)t * Sd, CC * Bn * Nn, Sd, Sd, 1);
    }
    eval_k<<<dim3((CC * Bn + 255) / 256), 256, 0, stream>>>(se_bf, w_ev, b_ev, out, c0);
  }
}
